// QAQAttent_61357902790863
// MI455X (gfx1250) — matrix operand order repaired and checked by reading, not compiled, not run on hardware
//
#include <hip/hip_runtime.h>
#include <hip/hip_bf16.h>

#define N_TURNS 384
#define T_SEQ   384
#define H_DIM   768
#define D_DIM   128
#define NT      (N_TURNS * T_SEQ)    // 147456
#define NN      (N_TURNS * N_TURNS)  // 147456
#define RSQRT_D 0.08838834764831845f

typedef __attribute__((ext_vector_type(16))) __bf16 v16bf;
typedef __attribute__((ext_vector_type(8)))  float  v8f;

union BFrag {
  v16bf v;
  unsigned int u[8];
  unsigned short h[16];
  uint4 q[2];
};

static __device__ __forceinline__ unsigned short f2bf(float f) {
  unsigned int u = __float_as_uint(f);
  u += 0x7FFFu + ((u >> 16) & 1u);   // round-to-nearest-even
  return (unsigned short)(u >> 16);
}
static __device__ __forceinline__ unsigned int pack2(float lo, float hi) {
  return (unsigned int)f2bf(lo) | ((unsigned int)f2bf(hi) << 16);
}

// A fragment: 16x32 bf16, rows row0..row0+15 of row-major src (row stride rs
// elements), cols cbase..cbase+31. ISA §7.12.2 layout; pairs contiguous ->
// two 16B loads per lane. rs and cbase must be multiples of 8.
static __device__ __forceinline__ BFrag load_a(const unsigned short* src,
                                               int row0, long rs, long cbase) {
  int lane = threadIdx.x & 31;
  int m = lane & 15;
  int kb = (lane & 16) ? 8 : 0;
  const unsigned short* p = src + (long)(row0 + m) * rs + cbase + kb;
  BFrag f;
  f.q[0] = *(const uint4*)(p);
  f.q[1] = *(const uint4*)(p + 16);
  return f;
}

// B fragment where B = M^T and M is row-major [Ncols][K]: lane column n maps
// to row (col0+n) of M; per-lane K-run contiguous -> two 16B loads.
static __device__ __forceinline__ BFrag load_b_transposed(const unsigned short* src,
                                                          int col0, long rs, long kbase) {
  int lane = threadIdx.x & 31;
  int n = lane & 15;
  int kb = (lane & 16) ? 8 : 0;
  const unsigned short* p = src + (long)(col0 + n) * rs + kbase + kb;
  BFrag f;
  f.q[0] = *(const uint4*)(p);
  f.q[1] = *(const uint4*)(p + 16);
  return f;
}

static __device__ __forceinline__ v8f wmma_bf16(const BFrag& a, const BFrag& b, v8f c) {
  return __builtin_amdgcn_wmma_f32_16x16x32_bf16(false, a.v, false, b.v,
                                                 (short)0, c, false, false);
}

// ---------------- kernel 1: x fp32 -> bf16 [N*T, H] ----------------
__global__ void cast_x_kernel(const float* __restrict__ x, unsigned short* __restrict__ xb) {
  long base = ((long)blockIdx.x * 256 + threadIdx.x) * 8;
  const float4* xp = (const float4*)(x + base);
  float4 a = xp[0], b = xp[1];
  unsigned int* o = (unsigned int*)(xb + base);
  o[0] = pack2(a.x, a.y); o[1] = pack2(a.z, a.w);
  o[2] = pack2(b.x, b.y); o[3] = pack2(b.z, b.w);
}

// ------- kernel 2: fuse Wq|Wk -> column-major bf16 [256][768], bias [256] ---
__global__ void prep_w_kernel(const float* __restrict__ Wq, const float* __restrict__ bq,
                              const float* __restrict__ Wk, const float* __restrict__ bk,
                              unsigned short* __restrict__ wcbT, float* __restrict__ bc) {
  int gid = blockIdx.x * 256 + threadIdx.x;  // < 98304, gid = h*128 + d
  int h = gid >> 7;
  int d = gid & 127;
  wcbT[(long)d * H_DIM + h]         = f2bf(Wq[gid]);
  wcbT[(long)(128 + d) * H_DIM + h] = f2bf(Wk[gid]);
  if (gid < 128) bc[gid] = bq[gid];
  else if (gid < 256) bc[gid] = bk[gid - 128];
}

// ------- kernel 3: transpose xb [j][t][h] -> xbt [t][h][j] via LDS tile -----
__global__ void transpose_x_kernel(const unsigned short* __restrict__ xb,
                                   unsigned short* __restrict__ xbt) {
  __shared__ unsigned short tile[64][65];
  int t  = blockIdx.x;
  int h0 = blockIdx.y * 64;
  int j0 = blockIdx.z * 64;
  int tx = threadIdx.x & 15;
  int ty = threadIdx.x >> 4;
#pragma unroll
  for (int rr = 0; rr < 4; ++rr) {
    int jl = ty * 4 + rr;
    const unsigned short* p = xb + ((long)(j0 + jl) * T_SEQ + t) * H_DIM + h0 + tx * 4;
    uint2 v = *(const uint2*)p;
    tile[jl][tx * 4 + 0] = (unsigned short)(v.x & 0xffffu);
    tile[jl][tx * 4 + 1] = (unsigned short)(v.x >> 16);
    tile[jl][tx * 4 + 2] = (unsigned short)(v.y & 0xffffu);
    tile[jl][tx * 4 + 3] = (unsigned short)(v.y >> 16);
  }
  __syncthreads();
#pragma unroll
  for (int rr = 0; rr < 4; ++rr) {
    int hl = ty * 4 + rr;
    unsigned short e0 = tile[tx * 4 + 0][hl];
    unsigned short e1 = tile[tx * 4 + 1][hl];
    unsigned short e2 = tile[tx * 4 + 2][hl];
    unsigned short e3 = tile[tx * 4 + 3][hl];
    uint2 v;
    v.x = (unsigned)e0 | ((unsigned)e1 << 16);
    v.y = (unsigned)e2 | ((unsigned)e3 << 16);
    *(uint2*)(xbt + (long)t * H_DIM * N_TURNS + (long)(h0 + hl) * N_TURNS + j0 + tx * 4) = v;
  }
}

// ------- kernel 4: [NT x 768] @ [768 x 256] + bias -> q,k bf16 --------------
// One wave computes a 16x64 strip; K-loop software-pipelined (prefetch kk+32).
__global__ void proj_kernel(const unsigned short* __restrict__ xb,
                            const unsigned short* __restrict__ wcbT,
                            const float* __restrict__ bc,
                            unsigned short* __restrict__ qb,
                            unsigned short* __restrict__ kb) {
  int wave = blockIdx.x * 8 + (threadIdx.x >> 5);  // 0..36863
  int it = wave >> 2;
  int colg = (wave & 3) * 64;
  int row0 = it * 16;
  v8f c0 = {}, c1 = {}, c2 = {}, c3 = {};
  BFrag a  = load_a(xb, row0, H_DIM, 0);
  BFrag b0 = load_b_transposed(wcbT, colg +  0, H_DIM, 0);
  BFrag b1 = load_b_transposed(wcbT, colg + 16, H_DIM, 0);
  BFrag b2 = load_b_transposed(wcbT, colg + 32, H_DIM, 0);
  BFrag b3 = load_b_transposed(wcbT, colg + 48, H_DIM, 0);
  for (int kk = 32; kk < H_DIM; kk += 32) {
    BFrag an  = load_a(xb, row0, H_DIM, kk);
    BFrag bn0 = load_b_transposed(wcbT, colg +  0, H_DIM, kk);
    BFrag bn1 = load_b_transposed(wcbT, colg + 16, H_DIM, kk);
    BFrag bn2 = load_b_transposed(wcbT, colg + 32, H_DIM, kk);
    BFrag bn3 = load_b_transposed(wcbT, colg + 48, H_DIM, kk);
    c0 = wmma_bf16(a, b0, c0);
    c1 = wmma_bf16(a, b1, c1);
    c2 = wmma_bf16(a, b2, c2);
    c3 = wmma_bf16(a, b3, c3);
    a = an; b0 = bn0; b1 = bn1; b2 = bn2; b3 = bn3;
  }
  c0 = wmma_bf16(a, b0, c0);
  c1 = wmma_bf16(a, b1, c1);
  c2 = wmma_bf16(a, b2, c2);
  c3 = wmma_bf16(a, b3, c3);

  int lane = threadIdx.x & 31;
  int n = lane & 15;
  int mb = (lane & 16) ? 8 : 0;
  v8f acc[4] = {c0, c1, c2, c3};
#pragma unroll
  for (int u = 0; u < 4; ++u) {
    int col = colg + 16 * u + n;
    float bias = bc[col];
#pragma unroll
    for (int r = 0; r < 8; ++r) {
      float val = acc[u][r] + bias;
      long row = row0 + mb + r;
      if (col < 128) qb[row * D_DIM + col] = f2bf(val);
      else           kb[row * D_DIM + (col - 128)] = f2bf(val);
    }
  }
}

// ------- kernel 5: per-t scores, lower-triangular tiles ---------------------
__global__ void scores_kernel(const unsigned short* __restrict__ qb,
                              const unsigned short* __restrict__ kbuf,
                              float* __restrict__ s) {
  int widx = blockIdx.x * 8 + (threadIdx.x >> 5);
  if (widx >= 300) return;  // 24*25/2 lower tiles, wave-uniform exit
  int t = blockIdx.y;
  int it = (int)((sqrtf(8.0f * (float)widx + 1.0f) - 1.0f) * 0.5f);
  while ((it + 1) * (it + 2) / 2 <= widx) ++it;
  while (it * (it + 1) / 2 > widx) --it;
  int jt = widx - it * (it + 1) / 2;
  int row0 = it * 16, col0 = jt * 16;
  long rs = (long)T_SEQ * D_DIM;
  long cb = (long)t * D_DIM;
  v8f c = {};
#pragma unroll
  for (int kk = 0; kk < D_DIM; kk += 32) {
    BFrag a = load_a(qb, row0, rs, cb + kk);
    BFrag b = load_b_transposed(kbuf, col0, rs, cb + kk);
    c = wmma_bf16(a, b, c);
  }
  int lane = threadIdx.x & 31;
  int n = lane & 15;
  int mb = (lane & 16) ? 8 : 0;
  float* sp = s + (long)t * NN;
#pragma unroll
  for (int r = 0; r < 8; ++r) {
    int i = row0 + mb + r;
    int j = col0 + n;
    float val = c[r] * RSQRT_D;
    if (j > i) val = -__builtin_inff();  // causal mask (diagonal tile)
    sp[(long)i * N_TURNS + j] = val;
  }
}

// ------- kernel 6: per-row joint softmax max & sum --------------------------
__global__ void reduce_kernel(const float* __restrict__ s,
                              float* __restrict__ mbuf, float* __restrict__ lbuf) {
  __shared__ float red[256];
  int i = blockIdx.x;
  int tid = threadIdx.x;
  float mx = -3.0e38f;
  for (int t = 0; t < T_SEQ; ++t) {
    const float* row = s + (long)t * NN + (long)i * N_TURNS;
    for (int j = tid; j <= i; j += 256) mx = fmaxf(mx, row[j]);
  }
  red[tid] = mx;
  __syncthreads();
  for (int off = 128; off > 0; off >>= 1) {
    if (tid < off) red[tid] = fmaxf(red[tid], red[tid + off]);
    __syncthreads();
  }
  float mi = red[0];
  __syncthreads();
  float sum = 0.0f;
  for (int t = 0; t < T_SEQ; ++t) {
    const float* row = s + (long)t * NN + (long)i * N_TURNS;
    for (int j = tid; j <= i; j += 256) sum += __expf(row[j] - mi);
  }
  red[tid] = sum;
  __syncthreads();
  for (int off = 128; off > 0; off >>= 1) {
    if (tid < off) red[tid] += red[tid + off];
    __syncthreads();
  }
  if (tid == 0) { mbuf[i] = mi; lbuf[i] = red[0]; }
}

// ------- kernel 7: alpha = exp(s - m)/l -> bf16 (0 above diagonal) ----------
__global__ void normalize_kernel(const float* __restrict__ s,
                                 const float* __restrict__ mbuf,
                                 const float* __restrict__ lbuf,
                                 unsigned short* __restrict__ ab) {
  int t = blockIdx.y;
  long base = ((long)blockIdx.x * 256 + threadIdx.x) * 8;  // < NN, 8 same-row j
  int i = (int)(base / N_TURNS);
  int j0 = (int)(base % N_TURNS);
  float mi = mbuf[i];
  float inv = 1.0f / lbuf[i];
  const float* sp = s + (long)t * NN + base;
  unsigned int* op = (unsigned int*)(ab + (long)t * NN + base);
  float v[8];
#pragma unroll
  for (int e = 0; e < 8; ++e) {
    int j = j0 + e;
    v[e] = (j <= i) ? __expf(sp[e] - mi) * inv : 0.0f;
  }
#pragma unroll
  for (int e = 0; e < 4; ++e) op[e] = pack2(v[2 * e], v[2 * e + 1]);
}

// ------- kernel 8: out_t = alpha_t @ x_t, 16x64 per wave, causal K,
//         software-pipelined ------------------------------------------------
__global__ void out_gemm_kernel(const unsigned short* __restrict__ ab,
                                const unsigned short* __restrict__ xbt,
                                float* __restrict__ out) {
  int wave = blockIdx.x * 8 + (threadIdx.x >> 5);  // 0..287 (24 it x 12 hg)
  int t = blockIdx.y;
  int it = wave / 12;
  int h0 = (wave % 12) * 64;
  int row0 = it * 16;
  const unsigned short* ap = ab + (long)t * NN;
  const unsigned short* bt = xbt + (long)t * H_DIM * N_TURNS;
  v8f c0 = {}, c1 = {}, c2 = {}, c3 = {};
  int kend = row0 + 16;  // alpha[i][j]=0 for j>i: skip upper K blocks
  BFrag a  = load_a(ap, row0, N_TURNS, 0);
  BFrag b0 = load_b_transposed(bt, h0 +  0, N_TURNS, 0);
  BFrag b1 = load_b_transposed(bt, h0 + 16, N_TURNS, 0);
  BFrag b2 = load_b_transposed(bt, h0 + 32, N_TURNS, 0);
  BFrag b3 = load_b_transposed(bt, h0 + 48, N_TURNS, 0);
  for (int kk = 32; kk < kend; kk += 32) {
    BFrag an  = load_a(ap, row0, N_TURNS, kk);
    BFrag bn0 = load_b_transposed(bt, h0 +  0, N_TURNS, kk);
    BFrag bn1 = load_b_transposed(bt, h0 + 16, N_TURNS, kk);
    BFrag bn2 = load_b_transposed(bt, h0 + 32, N_TURNS, kk);
    BFrag bn3 = load_b_transposed(bt, h0 + 48, N_TURNS, kk);
    c0 = wmma_bf16(a, b0, c0);
    c1 = wmma_bf16(a, b1, c1);
    c2 = wmma_bf16(a, b2, c2);
    c3 = wmma_bf16(a, b3, c3);
    a = an; b0 = bn0; b1 = bn1; b2 = bn2; b3 = bn3;
  }
  c0 = wmma_bf16(a, b0, c0);
  c1 = wmma_bf16(a, b1, c1);
  c2 = wmma_bf16(a, b2, c2);
  c3 = wmma_bf16(a, b3, c3);

  int lane = threadIdx.x & 31;
  int n = lane & 15;
  int mb = (lane & 16) ? 8 : 0;
  v8f acc[4] = {c0, c1, c2, c3};
#pragma unroll
  for (int u = 0; u < 4; ++u) {
#pragma unroll
    for (int r = 0; r < 8; ++r) {
      long row = row0 + mb + r;
      out[(row * T_SEQ + t) * (long)H_DIM + h0 + 16 * u + n] = acc[u][r];
    }
  }
}

// ------- kernel 9: row 0 pass-through ---------------------------------------
__global__ void row0_kernel(const float* __restrict__ x, float* __restrict__ out) {
  long base = ((long)blockIdx.x * 256 + threadIdx.x) * 8;  // < T*H
  const float4* xp = (const float4*)(x + base);
  float4* op = (float4*)(out + base);
  op[0] = xp[0];
  op[1] = xp[1];
}

extern "C" void kernel_launch(void* const* d_in, const int* in_sizes, int n_in,
                              void* d_out, int out_size, void* d_ws, size_t ws_size,
                              hipStream_t stream) {
  (void)in_sizes; (void)n_in; (void)out_size; (void)ws_size;
  const float* x  = (const float*)d_in[0];
  const float* Wq = (const float*)d_in[1];
  const float* bq = (const float*)d_in[2];
  const float* Wk = (const float*)d_in[3];
  const float* bk = (const float*)d_in[4];
  float* out = (float*)d_out;

  char* ws = (char*)d_ws;
  size_t off = 0;
  auto take = [&](size_t bytes) {
    char* p = ws + off;
    off = (off + bytes + 255) & ~(size_t)255;
    return p;
  };
  unsigned short* xb   = (unsigned short*)take((size_t)NT * H_DIM * 2);              // 226 MB
  unsigned short* xbt  = (unsigned short*)take((size_t)T_SEQ * H_DIM * N_TURNS * 2); // 226 MB
  unsigned short* wcbT = (unsigned short*)take((size_t)256 * H_DIM * 2);
  float*          bc   = (float*)take(256 * 4);
  unsigned short* qb   = (unsigned short*)take((size_t)NT * D_DIM * 2);              // 38 MB
  unsigned short* kb   = (unsigned short*)take((size_t)NT * D_DIM * 2);              // 38 MB
  float*          s    = (float*)take((size_t)T_SEQ * NN * 4);                       // 226 MB
  unsigned short* ab   = (unsigned short*)take((size_t)T_SEQ * NN * 2);              // 113 MB
  float*          mb   = (float*)take(N_TURNS * 4);
  float*          lb   = (float*)take(N_TURNS * 4);

  cast_x_kernel     <<<dim3(55296),      dim3(256), 0, stream>>>(x, xb);
  prep_w_kernel     <<<dim3(384),        dim3(256), 0, stream>>>(Wq, bq, Wk, bk, wcbT, bc);
  transpose_x_kernel<<<dim3(384, 12, 6), dim3(256), 0, stream>>>(xb, xbt);
  proj_kernel       <<<dim3(4608),       dim3(256), 0, stream>>>(xb, wcbT, bc, qb, kb);
  scores_kernel     <<<dim3(38, 384),    dim3(256), 0, stream>>>(qb, kb, s);
  reduce_kernel     <<<dim3(384),        dim3(256), 0, stream>>>(s, mb, lb);
  normalize_kernel  <<<dim3(72, 384),    dim3(256), 0, stream>>>(s, mb, lb, ab);
  out_gemm_kernel   <<<dim3(36, 384),    dim3(256), 0, stream>>>(ab, xbt, out);
  row0_kernel       <<<dim3(144),        dim3(256), 0, stream>>>(x, out);
}
